// AtomsGPT_37263136260137
// MI455X (gfx1250) — matrix operand order repaired and checked by reading, not compiled, not run on hardware
//
#include <hip/hip_runtime.h>
#include <hip/hip_bf16.h>
#include <math.h>

// ---- model dims (match reference) ----
#define B_  4
#define T_  1024
#define D_  1024
#define H_  16
#define HD_ 64
#define L_  8
#define V_  50257
#define BT_ (B_ * T_)

typedef __bf16 bf16_t;
typedef __attribute__((ext_vector_type(16))) __bf16 v16bf;
typedef __attribute__((ext_vector_type(8)))  float  v8f;
typedef __attribute__((ext_vector_type(4)))  unsigned int v4u_;
typedef __attribute__((ext_vector_type(8)))  int v8i_;
typedef __attribute__((ext_vector_type(4)))  int v4i_;

union AFrag { v16bf v; unsigned u[8]; };

__device__ __forceinline__ v8f wmma_bf16f32(v16bf a, v16bf b, v8f c) {
  // D = A(16x32 bf16) * B(32x16 bf16) + C(16x16 f32)
  return __builtin_amdgcn_wmma_f32_16x16x32_bf16(false, a, false, b, (short)0, c,
                                                 false, false);
}

__device__ __forceinline__ bf16_t to_bf16(float f) { return (bf16_t)f; }

// ---------------- Tensor Data Mover: 2D bf16 tile -> LDS ----------------
// Builds a D# (ISA 08_async_tensor §8.3/8.4) for a 2D tile of 2-byte elements:
//   group0: count=1, lds_addr, global tile address, type=2
//   group1: data_size=1(2B), tensor_dim0/1 (OOB zero-fill), tile_dim0/1,
//           tensor_dim0_stride (row stride, elements); groups 2/3 zero (2D).
__device__ __forceinline__ void tdm_load_tile_2d(unsigned lds_byte_addr,
                                                 const void* global_tile,
                                                 unsigned tensor_d0,
                                                 unsigned tensor_d1,
                                                 unsigned tile_d0,
                                                 unsigned tile_d1,
                                                 unsigned stride_d0) {
  const unsigned long long ga = (unsigned long long)global_tile;
  v4u_ g0;
  g0.x = 1u;                                     // count=1, user descriptor
  g0.y = lds_byte_addr;                          // D#.lds_addr [63:32]
  g0.z = (unsigned)(ga & 0xffffffffull);         // global addr [31:0]
  g0.w = (unsigned)((ga >> 32) & 0x1ffffffull)   // global addr [56:32]
         | (2u << 30);                           // type = 2 ("image")
  v8i_ g1;
  g1[0] = (int)(1u << 16);                       // data_size=1 -> 2 bytes
  g1[1] = (int)((tensor_d0 & 0xffffu) << 16);    // tensor_dim0 [15:0]
  g1[2] = (int)(((tensor_d0 >> 16) & 0xffffu) |  // tensor_dim0 [31:16]
                ((tensor_d1 & 0xffffu) << 16));  // tensor_dim1 [15:0]
  g1[3] = (int)(((tensor_d1 >> 16) & 0xffffu) |  // tensor_dim1 [31:16]
                ((tile_d0 & 0xffffu) << 16));    // tile_dim0
  g1[4] = (int)(tile_d1 & 0xffffu);              // tile_dim1 (tile_dim2=0)
  g1[5] = (int)stride_d0;                        // tensor_dim0_stride [31:0]
  g1[6] = 0;                                     // stride hi, dim1_stride=0
  g1[7] = 0;
  const v4i_ gz = {0, 0, 0, 0};
#if __clang_major__ >= 23
  const v8i_ gz8 = {0, 0, 0, 0, 0, 0, 0, 0};
  __builtin_amdgcn_tensor_load_to_lds(g0, g1, gz, gz, gz8, 0);
#else
  __builtin_amdgcn_tensor_load_to_lds(g0, g1, gz, gz, 0);
#endif
}

// A-fragment loader: row-major [rows, ld] bf16 source (LDS).
// ISA 16-bit A 16x32 layout: lane l (m = l&15, half = l>>4) holds
// K = {0..7}+8*half in VGPR0..3 (pairs), K = {16..23}+8*half in VGPR4..7.
__device__ __forceinline__ v16bf load_a_frag(const bf16_t* base, int ld, int row,
                                             int lane) {
  const int half = lane >> 4, m = lane & 15;
  AFrag f;
  const bf16_t* rp = base + (size_t)(row + m) * ld;
#pragma unroll
  for (int v = 0; v < 8; ++v) {
    const int kb = ((v < 4) ? (2 * v) : (16 + 2 * (v - 4))) + 8 * half;
    f.u[v] = *(const unsigned*)(rp + kb);  // pair K=kb,kb+1 (kb even -> aligned)
  }
  return f.v;
}

// B-fragment loader: row-major [K, ld] bf16 source; col = N column of tile.
// B 32x16 layout matching A: lane half h holds K = {0..7}+8*h in VGPR0..3,
// K = {16..23}+8*h in VGPR4..7 (pairs), for column n = lane&15.
__device__ __forceinline__ v16bf load_b_frag(const bf16_t* base, int ld, int col,
                                             int lane) {
  const int half = lane >> 4, n = lane & 15;
  AFrag f;
  const bf16_t* cp = base + (col + n);
#pragma unroll
  for (int v = 0; v < 8; ++v) {
    const int kb = ((v < 4) ? (2 * v) : (16 + 2 * (v - 4))) + 8 * half;
    const bf16_t* p0 = cp + (size_t)kb * ld;
    const unsigned lo = *(const unsigned short*)p0;
    const unsigned hi = *(const unsigned short*)(p0 + ld);
    f.u[v] = lo | (hi << 16);
  }
  return f.v;
}

// ---------------- GEMM: C[M,N] = A[M,K](bf16) x B[K,N](bf16) (+bias)(+gelu)(+resid)
// block = 256 threads = 8 waves; tile 128(M) x 64(N) x 32(K).
// TDM double-buffered tile staging: wave 0 issues tensor_load_to_lds for tile
// ks+1 while all 8 waves run WMMA on tile ks. OOB rows/cols zero-filled by TDM
// (handles the ragged N=50257 LM-head edge).
template <bool BIAS, bool GELU, bool RESID, bool OUTBF>
__global__ __launch_bounds__(256) void gemm_bf16_kernel(
    const bf16_t* __restrict__ A, const bf16_t* __restrict__ Bm,
    const float* __restrict__ bias, const float* __restrict__ resid,
    float* __restrict__ Cf, bf16_t* __restrict__ Cb, int M, int N, int K) {
  __shared__ bf16_t As[2 * 128 * 32];
  __shared__ bf16_t Bs[2 * 32 * 64];

  const int tid  = threadIdx.x;
  const int lane = tid & 31;
  const int wave = tid >> 5;
  const int wm = wave & 3, wn = wave >> 2;
  const int row0 = blockIdx.y * 128;
  const int col0 = blockIdx.x * 64;
  const int half = lane >> 4, n = lane & 15;

  const unsigned ldsA = (unsigned)(size_t)As;
  const unsigned ldsB = (unsigned)(size_t)Bs;

  v8f acc[2][2];
  const v8f vzero = {0.f, 0.f, 0.f, 0.f, 0.f, 0.f, 0.f, 0.f};
  acc[0][0] = vzero; acc[0][1] = vzero; acc[1][0] = vzero; acc[1][1] = vzero;

  const int ksteps = K >> 5;
  if (wave == 0) {  // prologue: DMA tile 0 into buffer 0
    tdm_load_tile_2d(ldsA, A + (size_t)row0 * K, (unsigned)K,
                     (unsigned)(M - row0), 32u, 128u, (unsigned)K);
    tdm_load_tile_2d(ldsB, Bm + col0, (unsigned)(N - col0), (unsigned)K, 64u,
                     32u, (unsigned)N);
  }
  for (int ks = 0; ks < ksteps; ++ks) {
    const int cur = ks & 1;
    if (wave == 0) __builtin_amdgcn_s_wait_tensorcnt(0);
    __syncthreads();  // tile ks visible to all waves
    if (wave == 0 && ks + 1 < ksteps) {  // DMA tile ks+1 into other buffer
      const int k1 = (ks + 1) << 5;
      const int nxt = cur ^ 1;
      tdm_load_tile_2d(ldsA + (unsigned)nxt * (128 * 32 * 2),
                       A + (size_t)row0 * K + k1, (unsigned)(K - k1),
                       (unsigned)(M - row0), 32u, 128u, (unsigned)K);
      tdm_load_tile_2d(ldsB + (unsigned)nxt * (32 * 64 * 2),
                       Bm + (size_t)k1 * N + col0, (unsigned)(N - col0),
                       (unsigned)(K - k1), 64u, 32u, (unsigned)N);
    }
    const bf16_t* Ab = As + cur * (128 * 32);
    const bf16_t* Bb = Bs + cur * (32 * 64);

    v16bf af[2], bfm[2];
#pragma unroll
    for (int tm = 0; tm < 2; ++tm)
      af[tm] = load_a_frag(Ab, 32, wm * 32 + tm * 16, lane);
#pragma unroll
    for (int tn = 0; tn < 2; ++tn)
      bfm[tn] = load_b_frag(Bb, 64, wn * 32 + tn * 16, lane);
#pragma unroll
    for (int tm = 0; tm < 2; ++tm)
#pragma unroll
      for (int tn = 0; tn < 2; ++tn)
        acc[tm][tn] = wmma_bf16f32(af[tm], bfm[tn], acc[tm][tn]);
    __syncthreads();  // all reads of tile ks done before its buffer is reused
  }

  // epilogue: C/D layout -> lane (half,n), VGPR r holds row r+8*half, col n
#pragma unroll
  for (int tm = 0; tm < 2; ++tm)
#pragma unroll
    for (int tn = 0; tn < 2; ++tn)
#pragma unroll
      for (int r = 0; r < 8; ++r) {
        const int grow = row0 + wm * 32 + tm * 16 + r + 8 * half;
        const int gcol = col0 + wn * 32 + tn * 16 + n;
        if (grow < M && gcol < N) {
          float v = acc[tm][tn][r];
          if (BIAS)  v += bias[gcol];
          if (GELU)  v = 0.5f * v * (1.0f + erff(v * 0.70710678118654752f));
          if (RESID) v += resid[(size_t)grow * N + gcol];
          if (OUTBF) Cb[(size_t)grow * N + gcol] = to_bf16(v);
          else       Cf[(size_t)grow * N + gcol] = v;
        }
      }
}

// ---------------- Flash attention, causal, scale=1/8 (HD=64) ----------------
// grid = (B*H, T/128); block = 256 = 8 waves; wave w owns 16 query rows.
// K/V tiles (32 keys x 64 dims) DMA'd to LDS via TDM by wave 0; S and PV via
// WMMA; per-wave LDS staging of P re-layouts C->A fragments (DScnt wait only).
__global__ __launch_bounds__(256) void attention_kernel(
    const bf16_t* __restrict__ qkv, bf16_t* __restrict__ out) {
  __shared__ bf16_t Ks[32 * 64];
  __shared__ bf16_t Vs[32 * 64];
  __shared__ bf16_t Ps[8][16 * 32];

  const int tid = threadIdx.x, lane = tid & 31, wave = tid >> 5;
  const int bh = blockIdx.x, b = bh / H_, h = bh % H_;
  const int qblock = blockIdx.y * 128;
  const int q0 = qblock + wave * 16;
  const int half = lane >> 4, n = lane & 15;
  const int rowStride = 3 * D_;

  const unsigned ldsK = (unsigned)(size_t)Ks;
  const unsigned ldsV = (unsigned)(size_t)Vs;

  // Q fragments for this wave's 16 rows: K-dim 64 -> two 16x32 A fragments
  v16bf qa[2];
  {
    AFrag f0, f1;
    const int m = lane & 15;
    const bf16_t* rp = qkv + (size_t)(b * T_ + q0 + m) * rowStride + h * HD_;
#pragma unroll
    for (int v = 0; v < 8; ++v) {
      const int kb = ((v < 4) ? (2 * v) : (16 + 2 * (v - 4))) + 8 * half;
      f0.u[v] = *(const unsigned*)(rp + kb);
      f1.u[v] = *(const unsigned*)(rp + 32 + kb);
    }
    qa[0] = f0.v; qa[1] = f1.v;
  }

  const v8f vzero = {0.f, 0.f, 0.f, 0.f, 0.f, 0.f, 0.f, 0.f};
  v8f acc[4];
#pragma unroll
  for (int i = 0; i < 4; ++i) acc[i] = vzero;
  float mrow[8], lrow[8];
#pragma unroll
  for (int r = 0; r < 8; ++r) { mrow[r] = -1e30f; lrow[r] = 0.f; }

  const int nkt = qblock / 32 + 4;  // causal limit for this block
  for (int kt = 0; kt < nkt; ++kt) {
    if (wave == 0) {  // TDM both K and V 32x64 tiles, then drain TENSORcnt
      const bf16_t* kb =
          qkv + (size_t)(b * T_ + kt * 32) * rowStride + D_ + h * HD_;
      const bf16_t* vb =
          qkv + (size_t)(b * T_ + kt * 32) * rowStride + 2 * D_ + h * HD_;
      tdm_load_tile_2d(ldsK, kb, 64u, 32u, 64u, 32u, (unsigned)rowStride);
      tdm_load_tile_2d(ldsV, vb, 64u, 32u, 64u, 32u, (unsigned)rowStride);
      __builtin_amdgcn_s_wait_tensorcnt(0);
    }
    __syncthreads();  // tiles visible

    if (kt * 32 <= q0 + 15) {  // wave-uniform causal participation
      // S = Q K^T : two 16(key)-wide tiles, contraction over HD=64 in 2 steps
      v8f s[2];
#pragma unroll
      for (int j = 0; j < 2; ++j) {
        v8f c = vzero;
#pragma unroll
        for (int kk = 0; kk < 2; ++kk) {
          AFrag bk;
#pragma unroll
          for (int v = 0; v < 8; ++v) {
            const int krow = kk * 32 + ((v < 4) ? (2 * v) : (16 + 2 * (v - 4))) + 8 * half;  // head-dim (even)
            bk.u[v] = *(const unsigned*)(&Ks[(j * 16 + n) * 64 + krow]);
          }
          c = wmma_bf16f32(qa[kk], bk.v, c);
        }
        s[j] = c;
      }

      // online softmax (rows r+8*half live across the 16 lanes of each half)
      const float scale = 0.125f;
#pragma unroll
      for (int r = 0; r < 8; ++r) {
        const int q = q0 + r + 8 * half;
        float c0 = s[0][r] * scale;
        float c1 = s[1][r] * scale;
        const int key0 = kt * 32 + n;
        if (key0 > q)      c0 = -1e30f;
        if (key0 + 16 > q) c1 = -1e30f;
        float cand = fmaxf(c0, c1);
        cand = fmaxf(cand, __shfl_xor(cand, 1, 32));
        cand = fmaxf(cand, __shfl_xor(cand, 2, 32));
        cand = fmaxf(cand, __shfl_xor(cand, 4, 32));
        cand = fmaxf(cand, __shfl_xor(cand, 8, 32));
        const float mnew = fmaxf(mrow[r], cand);
        const float sf = __expf(mrow[r] - mnew);
        const float p0 = __expf(c0 - mnew);
        const float p1 = __expf(c1 - mnew);
        float rs = p0 + p1;
        rs += __shfl_xor(rs, 1, 32);
        rs += __shfl_xor(rs, 2, 32);
        rs += __shfl_xor(rs, 4, 32);
        rs += __shfl_xor(rs, 8, 32);
        lrow[r] = lrow[r] * sf + rs;
        mrow[r] = mnew;
#pragma unroll
        for (int nt = 0; nt < 4; ++nt) acc[nt][r] *= sf;
        Ps[wave][(r + 8 * half) * 32 + n]      = to_bf16(p0);
        Ps[wave][(r + 8 * half) * 32 + 16 + n] = to_bf16(p1);
      }
      // wave-local staging: only need LDS ops of THIS wave retired
      asm volatile("s_wait_dscnt 0" ::: "memory");

      // reload P as an A fragment and accumulate O += P * V
      AFrag pf;
      {
        const int m = lane & 15;
#pragma unroll
        for (int v = 0; v < 8; ++v) {
          const int kb = ((v < 4) ? (2 * v) : (16 + 2 * (v - 4))) + 8 * half;
          pf.u[v] = *(const unsigned*)(&Ps[wave][m * 32 + kb]);
        }
      }
#pragma unroll
      for (int nt = 0; nt < 4; ++nt) {
        AFrag bv;
#pragma unroll
        for (int v = 0; v < 8; ++v) {
          const int krow = ((v < 4) ? (2 * v) : (16 + 2 * (v - 4))) + 8 * half;  // key index
          const unsigned lo =
              *(const unsigned short*)&Vs[krow * 64 + nt * 16 + n];
          const unsigned hi =
              *(const unsigned short*)&Vs[(krow + 1) * 64 + nt * 16 + n];
          bv.u[v] = lo | (hi << 16);
        }
        acc[nt] = wmma_bf16f32(pf.v, bv.v, acc[nt]);
      }
    }
    __syncthreads();  // reads done before next TDM overwrites Ks/Vs
  }

  // normalize and write O (bf16) into [BT, D] at columns h*64..h*64+63
#pragma unroll
  for (int nt = 0; nt < 4; ++nt)
#pragma unroll
    for (int r = 0; r < 8; ++r) {
      const int q = q0 + r + 8 * half;
      const int col = h * HD_ + nt * 16 + n;
      out[(size_t)(b * T_ + q) * D_ + col] = to_bf16(acc[nt][r] / lrow[r]);
    }
}

// ---------------- LayerNorm over D=1024, one block per row, bf16 out ----------
__global__ __launch_bounds__(256) void layernorm_kernel(
    const float* __restrict__ x, const float* __restrict__ s,
    const float* __restrict__ bta, bf16_t* __restrict__ out) {
  __shared__ float red[256];
  const int row = blockIdx.x, tid = threadIdx.x;
  const float* xr = x + (size_t)row * D_;
  const float4 v = *(const float4*)(xr + tid * 4);

  red[tid] = v.x + v.y + v.z + v.w;
  __syncthreads();
  for (int st = 128; st > 0; st >>= 1) {
    if (tid < st) red[tid] += red[tid + st];
    __syncthreads();
  }
  const float mean = red[0] * (1.0f / D_);
  __syncthreads();
  const float d0 = v.x - mean, d1 = v.y - mean, d2 = v.z - mean, d3 = v.w - mean;
  red[tid] = d0 * d0 + d1 * d1 + d2 * d2 + d3 * d3;
  __syncthreads();
  for (int st = 128; st > 0; st >>= 1) {
    if (tid < st) red[tid] += red[tid + st];
    __syncthreads();
  }
  const float rstd = rsqrtf(red[0] * (1.0f / D_) + 1e-5f);
  const int c = tid * 4;
  bf16_t* op = out + (size_t)row * D_ + c;
  op[0] = to_bf16(d0 * rstd * s[c + 0] + bta[c + 0]);
  op[1] = to_bf16(d1 * rstd * s[c + 1] + bta[c + 1]);
  op[2] = to_bf16(d2 * rstd * s[c + 2] + bta[c + 2]);
  op[3] = to_bf16(d3 * rstd * s[c + 3] + bta[c + 3]);
}

// ---------------- small elementwise kernels ----------------
__global__ void embed_kernel(const int* __restrict__ idx,
                             const float* __restrict__ tok,
                             const float* __restrict__ pos,
                             float* __restrict__ x) {
  const size_t i = (size_t)blockIdx.x * blockDim.x + threadIdx.x;
  if (i >= (size_t)BT_ * D_) return;
  const int bt = (int)(i / D_), d = (int)(i % D_);
  const int t = bt % T_;
  const int tk = idx[bt];
  x[i] = tok[(size_t)tk * D_ + d] + pos[(size_t)t * D_ + d];
}

__global__ void f32_to_bf16_kernel(const float* __restrict__ in,
                                   bf16_t* __restrict__ out, size_t n) {
  const size_t i = (size_t)blockIdx.x * blockDim.x + threadIdx.x;
  if (i < n) out[i] = to_bf16(in[i]);
}

__global__ void transpose_bf16_kernel(const float* __restrict__ in,
                                      bf16_t* __restrict__ out) {
  const size_t i = (size_t)blockIdx.x * blockDim.x + threadIdx.x;
  if (i >= (size_t)V_ * D_) return;
  const int v = (int)(i / D_), d = (int)(i % D_);
  out[(size_t)d * V_ + v] = to_bf16(in[i]);
}

// ---------------- driver ----------------
extern "C" void kernel_launch(void* const* d_in, const int* in_sizes, int n_in,
                              void* d_out, int out_size, void* d_ws,
                              size_t ws_size, hipStream_t stream) {
  (void)in_sizes; (void)n_in; (void)out_size; (void)ws_size;
  const int*   idx    = (const int*)d_in[0];
  const float* tokw   = (const float*)d_in[1];
  const float* posw   = (const float*)d_in[2];
  const float* qkv_w  = (const float*)d_in[3];
  const float* qkv_b  = (const float*)d_in[4];
  const float* proj_w = (const float*)d_in[5];
  const float* proj_b = (const float*)d_in[6];
  const float* ff1_w  = (const float*)d_in[7];
  const float* ff1_b  = (const float*)d_in[8];
  const float* ff2_w  = (const float*)d_in[9];
  const float* ff2_b  = (const float*)d_in[10];
  const float* ln1_s  = (const float*)d_in[11];
  const float* ln1_b  = (const float*)d_in[12];
  const float* ln2_s  = (const float*)d_in[13];
  const float* ln2_b  = (const float*)d_in[14];
  const float* lnf_s  = (const float*)d_in[15];
  const float* lnf_b  = (const float*)d_in[16];
  float* logits = (float*)d_out;

  char* ws = (char*)d_ws;
  size_t off = 0;
  auto alloc = [&](size_t bytes) -> char* {
    char* p = ws + off;
    off += (bytes + 255) & ~(size_t)255;
    return p;
  };
  bf16_t* wq   = (bf16_t*)alloc((size_t)L_ * D_ * 3 * D_ * 2);
  bf16_t* wp   = (bf16_t*)alloc((size_t)L_ * D_ * D_ * 2);
  bf16_t* wf1  = (bf16_t*)alloc((size_t)L_ * D_ * 4 * D_ * 2);
  bf16_t* wf2  = (bf16_t*)alloc((size_t)L_ * 4 * D_ * D_ * 2);
  bf16_t* embT = (bf16_t*)alloc((size_t)D_ * V_ * 2);
  float*  x    = (float*)alloc((size_t)BT_ * D_ * 4);
  bf16_t* hbf  = (bf16_t*)alloc((size_t)BT_ * D_ * 2);
  bf16_t* qkvb = (bf16_t*)alloc((size_t)BT_ * 3 * D_ * 2);
  bf16_t* attb = (bf16_t*)alloc((size_t)BT_ * D_ * 2);
  bf16_t* ffh  = (bf16_t*)alloc((size_t)BT_ * 4 * D_ * 2);

  auto cvt = [&](const float* src, bf16_t* dst, size_t nelem) {
    f32_to_bf16_kernel<<<dim3((unsigned)((nelem + 255) / 256)), 256, 0, stream>>>(
        src, dst, nelem);
  };
  cvt(qkv_w,  wq,  (size_t)L_ * D_ * 3 * D_);
  cvt(proj_w, wp,  (size_t)L_ * D_ * D_);
  cvt(ff1_w,  wf1, (size_t)L_ * D_ * 4 * D_);
  cvt(ff2_w,  wf2, (size_t)L_ * 4 * D_ * D_);
  transpose_bf16_kernel<<<dim3((unsigned)(((size_t)V_ * D_ + 255) / 256)), 256, 0,
                          stream>>>(tokw, embT);
  embed_kernel<<<dim3((BT_ * D_ + 255) / 256), 256, 0, stream>>>(idx, tokw, posw, x);

  for (int l = 0; l < L_; ++l) {
    layernorm_kernel<<<BT_, 256, 0, stream>>>(x, ln1_s + l * D_, ln1_b + l * D_, hbf);
    gemm_bf16_kernel<true, false, false, true>
        <<<dim3(3 * D_ / 64, BT_ / 128), 256, 0, stream>>>(
            hbf, wq + (size_t)l * D_ * 3 * D_, qkv_b + l * 3 * D_, nullptr,
            nullptr, qkvb, BT_, 3 * D_, D_);
    attention_kernel<<<dim3(B_ * H_, T_ / 128), 256, 0, stream>>>(qkvb, attb);
    gemm_bf16_kernel<true, false, true, false>
        <<<dim3(D_ / 64, BT_ / 128), 256, 0, stream>>>(
            attb, wp + (size_t)l * D_ * D_, proj_b + l * D_, x, x, nullptr,
            BT_, D_, D_);
    layernorm_kernel<<<BT_, 256, 0, stream>>>(x, ln2_s + l * D_, ln2_b + l * D_, hbf);
    gemm_bf16_kernel<true, true, false, true>
        <<<dim3(4 * D_ / 64, BT_ / 128), 256, 0, stream>>>(
            hbf, wf1 + (size_t)l * D_ * 4 * D_, ff1_b + l * 4 * D_, nullptr,
            nullptr, ffh, BT_, 4 * D_, D_);
    gemm_bf16_kernel<true, false, true, false>
        <<<dim3(D_ / 64, BT_ / 128), 256, 0, stream>>>(
            ffh, wf2 + (size_t)l * 4 * D_ * D_, ff2_b + l * D_, x, x, nullptr,
            BT_, D_, 4 * D_);
  }
  layernorm_kernel<<<BT_, 256, 0, stream>>>(x, lnf_s, lnf_b, hbf);
  gemm_bf16_kernel<false, false, false, false>
      <<<dim3((V_ + 63) / 64, BT_ / 128), 256, 0, stream>>>(
          hbf, embT, nullptr, nullptr, logits, nullptr, BT_, V_, D_);
}
